// AtlasLayerBlock_65395172049082
// MI455X (gfx1250) — compile-verified, matrix operand order checked by reading, not run on hardware
//
#include <hip/hip_runtime.h>
#include <hip/hip_bf16.h>

typedef __attribute__((ext_vector_type(16))) __bf16 v16bf;
typedef __attribute__((ext_vector_type(8)))  float  v8f;
typedef unsigned short u16t;
typedef unsigned int   u32t;

#define BB 4
#define SS 4096
#define DD 1024
#define NCH 64
#define CHK 64
#define WIN 128

static __device__ __forceinline__ u16t f2bf(float f) {
  u32t u = __float_as_uint(f);
  u = (u + 0x7FFFu + ((u >> 16) & 1u)) >> 16;   // round-to-nearest-even
  return (u16t)u;
}

static __device__ __forceinline__ float silu_f(float v) {
  return v / (1.0f + __expf(-v));
}

// global -> 4 named uint4 registers (issue all loads back-to-back)
template <bool TA, bool TB>
static __device__ __forceinline__ void tile_gload(
    const u16t* __restrict__ Ab, const u16t* __restrict__ Bb,
    int m0, int n0, int M, int lda, int ldb, int k0, int tid,
    uint4& a0, uint4& a1, uint4& b0, uint4& b1)
{
  if (!TA) {
    int mi = tid >> 2, ki = (tid & 3) * 8;
    int gm = m0 + mi; if (gm >= M) gm = M - 1;     // branchless clamp
    a0 = *reinterpret_cast<const uint4*>(Ab + (size_t)gm * lda + (k0 + ki));
    int idx = tid + 256;
    mi = idx >> 2; ki = (idx & 3) * 8;
    gm = m0 + mi; if (gm >= M) gm = M - 1;
    a1 = *reinterpret_cast<const uint4*>(Ab + (size_t)gm * lda + (k0 + ki));
  } else {
    int ki = tid >> 4, mo = (tid & 15) * 8;
    a0 = *reinterpret_cast<const uint4*>(Ab + (size_t)(k0 + ki) * lda + (m0 + mo));
    int idx = tid + 256;
    ki = idx >> 4; mo = (idx & 15) * 8;
    a1 = *reinterpret_cast<const uint4*>(Ab + (size_t)(k0 + ki) * lda + (m0 + mo));
  }
  if (TB) {
    int ni = tid >> 2, ki = (tid & 3) * 8;
    b0 = *reinterpret_cast<const uint4*>(Bb + (size_t)(n0 + ni) * ldb + (k0 + ki));
    int idx = tid + 256;
    ni = idx >> 2; ki = (idx & 3) * 8;
    b1 = *reinterpret_cast<const uint4*>(Bb + (size_t)(n0 + ni) * ldb + (k0 + ki));
  } else {
    int ki = tid >> 4, no = (tid & 15) * 8;
    b0 = *reinterpret_cast<const uint4*>(Bb + (size_t)(k0 + ki) * ldb + (n0 + no));
    int idx = tid + 256;
    ki = idx >> 4; no = (idx & 15) * 8;
    b1 = *reinterpret_cast<const uint4*>(Bb + (size_t)(k0 + ki) * ldb + (n0 + no));
  }
}

// 4 uint4 values (by value: pure SSA) -> LDS buffer
template <bool TA, bool TB>
static __device__ __forceinline__ void tile_lstore(
    u16t (* __restrict__ lA)[40], u16t (* __restrict__ lB)[40], int tid,
    uint4 a0, uint4 a1, uint4 b0, uint4 b1)
{
  if (!TA) {
    *reinterpret_cast<uint4*>(&lA[tid >> 2][(tid & 3) * 8]) = a0;
    int idx = tid + 256;
    *reinterpret_cast<uint4*>(&lA[idx >> 2][(idx & 3) * 8]) = a1;
  } else {
    union { uint4 q; u16t hh[8]; } o;
    int ki = tid >> 4, mo = (tid & 15) * 8;
    o.q = a0;
#pragma unroll
    for (int e = 0; e < 8; ++e) lA[mo + e][ki] = o.hh[e];
    int idx = tid + 256;
    ki = idx >> 4; mo = (idx & 15) * 8;
    o.q = a1;
#pragma unroll
    for (int e = 0; e < 8; ++e) lA[mo + e][ki] = o.hh[e];
  }
  if (TB) {
    *reinterpret_cast<uint4*>(&lB[tid >> 2][(tid & 3) * 8]) = b0;
    int idx = tid + 256;
    *reinterpret_cast<uint4*>(&lB[idx >> 2][(idx & 3) * 8]) = b1;
  } else {
    union { uint4 q; u16t hh[8]; } o;
    int ki = tid >> 4, no = (tid & 15) * 8;
    o.q = b0;
#pragma unroll
    for (int e = 0; e < 8; ++e) lB[no + e][ki] = o.hh[e];
    int idx = tid + 256;
    ki = idx >> 4; no = (idx & 15) * 8;
    o.q = b1;
#pragma unroll
    for (int e = 0; e < 8; ++e) lB[no + e][ki] = o.hh[e];
  }
}

// ---------------------------------------------------------------------------
// WMMA GEMM, bf16 operands, f32 accumulate.  D = epi( A(MxK) * B(KxN) )
//   TA: A stored (K x M) row-major; TB: B stored (N x K) row-major
//   EPI: 0 -> alpha*acc + (Cx? beta*Cx : 0);  1 -> silu(acc)
//   WF32 / WBF16: write f32 D and/or bf16 Dh
// Block tile 128x128, 256 threads = 8 wave32 at 32x64 (2x4 WMMA tiles each).
// Register-staged pipeline with NAMED uint4 values (no arrays/lambdas so
// mem2reg keeps them in VGPRs): next tile's global_load_b128 issue BEFORE the
// WMMA block and the LDS stores land AFTER it.
// ---------------------------------------------------------------------------
template <bool TA, bool TB, int EPI, bool WF32, bool WBF16>
__global__ __launch_bounds__(256, 1) void gemm_k(
    const u16t* __restrict__ A, const u16t* __restrict__ B,
    const float* __restrict__ Cx, float* __restrict__ D, u16t* __restrict__ Dh,
    int M, int N, int K, int lda, int ldb, int ldd,
    long sA, long sB, long sC, long sD, float alpha, float beta)
{
  __shared__ __align__(16) u16t lA[2][128][40];
  __shared__ __align__(16) u16t lB[2][128][40];

  const int bz = blockIdx.z;
  const u16t* Ab = A + (size_t)bz * (size_t)sA;
  const u16t* Bb = B + (size_t)bz * (size_t)sB;
  const float* Cb = Cx ? (Cx + (size_t)bz * (size_t)sC) : nullptr;
  float* Db  = D  ? (D  + (size_t)bz * (size_t)sD) : nullptr;
  u16t*  Dhb = Dh ? (Dh + (size_t)bz * (size_t)sD) : nullptr;

  const int tid  = threadIdx.x;
  const int lane = tid & 31;
  const int wave = tid >> 5;
  const int wm   = wave >> 1;   // 0..3  (32-row slabs)
  const int wn   = wave & 1;    // 0..1  (64-col slabs)
  const int h    = lane >> 4;
  const int ln   = lane & 15;

  const int m0 = blockIdx.y * 128;
  const int n0 = blockIdx.x * 128;

  v8f acc[2][4];
#pragma unroll
  for (int i = 0; i < 2; ++i)
#pragma unroll
    for (int j = 0; j < 4; ++j)
#pragma unroll
      for (int e = 0; e < 8; ++e) acc[i][j][e] = 0.0f;

  uint4 ta0, ta1, tb0, tb1;
  tile_gload<TA, TB>(Ab, Bb, m0, n0, M, lda, ldb, 0, tid, ta0, ta1, tb0, tb1);
  tile_lstore<TA, TB>(lA[0], lB[0], tid, ta0, ta1, tb0, tb1);
  __syncthreads();

  int p = 0;
  for (int k0 = 0; k0 < K; k0 += 32) {
    const bool nxt = (k0 + 32 < K);
    if (nxt)   // loads in flight during WMMAs
      tile_gload<TA, TB>(Ab, Bb, m0, n0, M, lda, ldb, k0 + 32, tid, ta0, ta1, tb0, tb1);

    v16bf af[2], bfr[4];
#pragma unroll
    for (int fi = 0; fi < 2; ++fi) {
      int row = wm * 32 + fi * 16 + ln;
      union { uint4 q[2]; v16bf v; } u;
      u.q[0] = *reinterpret_cast<const uint4*>(&lA[p][row][8 * h]);
      u.q[1] = *reinterpret_cast<const uint4*>(&lA[p][row][16 + 8 * h]);
      af[fi] = u.v;
    }
#pragma unroll
    for (int fj = 0; fj < 4; ++fj) {
      int row = wn * 64 + fj * 16 + ln;
      union { uint4 q[2]; v16bf v; } u;
      u.q[0] = *reinterpret_cast<const uint4*>(&lB[p][row][8 * h]);
      u.q[1] = *reinterpret_cast<const uint4*>(&lB[p][row][16 + 8 * h]);
      bfr[fj] = u.v;
    }
#pragma unroll
    for (int fi = 0; fi < 2; ++fi)
#pragma unroll
      for (int fj = 0; fj < 4; ++fj)
        acc[fi][fj] = __builtin_amdgcn_wmma_f32_16x16x32_bf16(
            false, af[fi], false, bfr[fj], (short)0, acc[fi][fj], false, false);

    if (nxt)   // waits land after the WMMA block
      tile_lstore<TA, TB>(lA[1 - p], lB[1 - p], tid, ta0, ta1, tb0, tb1);
    __syncthreads();
    p ^= 1;
  }

#pragma unroll
  for (int fi = 0; fi < 2; ++fi)
#pragma unroll
    for (int fj = 0; fj < 4; ++fj)
#pragma unroll
      for (int r = 0; r < 8; ++r) {
        int row = m0 + wm * 32 + fi * 16 + r + 8 * h;
        int col = n0 + wn * 64 + fj * 16 + ln;
        if (row < M) {
          float v = acc[fi][fj][r];
          if (EPI == 1) {
            v = silu_f(v);
          } else {
            float c = Cb ? beta * Cb[(size_t)row * ldd + col] : 0.0f;
            v = alpha * v + c;
          }
          if (WF32)  Db[(size_t)row * ldd + col]  = v;
          if (WBF16) Dhb[(size_t)row * ldd + col] = f2bf(v);
        }
      }
}

// ---------------------------------------------------------------------------
__global__ __launch_bounds__(256) void cvt_k(
    const float* __restrict__ s, u16t* __restrict__ d, int n)
{
  int i = blockIdx.x * 256 + threadIdx.x;
  if (i < n) d[i] = f2bf(s[i]);
}

// rmsnorm rows -> bf16 output (GEMM operand only)
__global__ __launch_bounds__(256) void rmsnorm_rows(
    const float* __restrict__ x, const float* __restrict__ w,
    u16t* __restrict__ y)
{
  __shared__ float red[256];
  size_t row = blockIdx.x;
  const float* xr = x + row * DD;
  u16t* yr = y + row * DD;
  float v[4], s = 0.0f;
#pragma unroll
  for (int i = 0; i < 4; ++i) { v[i] = xr[threadIdx.x + i * 256]; s += v[i] * v[i]; }
  red[threadIdx.x] = s; __syncthreads();
  for (int o = 128; o > 0; o >>= 1) {
    if (threadIdx.x < o) red[threadIdx.x] += red[threadIdx.x + o];
    __syncthreads();
  }
  float inv = rsqrtf(red[0] * (1.0f / DD) + 1e-6f);
#pragma unroll
  for (int i = 0; i < 4; ++i) {
    int d = threadIdx.x + i * 256;
    yr[d] = f2bf(v[i] * inv * w[d]);
  }
}

// depthwise conv3 + silu + rmsnorm; writes f32 and/or bf16
__global__ __launch_bounds__(256) void conv_silu_rms(
    const float* __restrict__ raw, const float* __restrict__ cw,
    const float* __restrict__ cb, const float* __restrict__ nw,
    float* __restrict__ outf, u16t* __restrict__ outh)
{
  __shared__ float red[256];
  int s = blockIdx.x, b = blockIdx.y;
  size_t ro = ((size_t)b * SS + s) * DD;
  const float* base = raw + ro;
  float v[4], acc = 0.0f;
#pragma unroll
  for (int i = 0; i < 4; ++i) {
    int d = threadIdx.x + i * 256;
    float xm = (s > 0)      ? base[(long)d - DD] : 0.0f;
    float xc = base[d];
    float xp = (s < SS - 1) ? base[(long)d + DD] : 0.0f;
    float y = cw[d * 3] * xm + cw[d * 3 + 1] * xc + cw[d * 3 + 2] * xp + cb[d];
    y = silu_f(y);
    v[i] = y; acc += y * y;
  }
  red[threadIdx.x] = acc; __syncthreads();
  for (int o = 128; o > 0; o >>= 1) {
    if (threadIdx.x < o) red[threadIdx.x] += red[threadIdx.x + o];
    __syncthreads();
  }
  float inv = rsqrtf(red[0] * (1.0f / DD) + 1e-6f);
#pragma unroll
  for (int i = 0; i < 4; ++i) {
    int d = threadIdx.x + i * 256;
    float y = v[i] * inv * nw[d];
    if (outf) outf[ro + d] = y;
    if (outh) outh[ro + d] = f2bf(y);
  }
}

__global__ __launch_bounds__(256) void chunk_mean(
    const float* __restrict__ t, float* __restrict__ out)
{
  int bc = blockIdx.x;
  int d = blockIdx.y * 256 + threadIdx.x;
  int b = bc >> 6, c = bc & 63;
  const float* p = t + ((size_t)b * SS + (size_t)c * CHK) * DD + d;
  float s = 0.0f;
  for (int i = 0; i < CHK; ++i) s += p[(size_t)i * DD];
  out[(size_t)bc * DD + d] = s * (1.0f / CHK);
}

// ctx_k (bf16) / ctx_v (f32) from ring buffer + chunk; zero normacc
__global__ __launch_bounds__(256) void build_ctx(
    const float* __restrict__ kbuf, const float* __restrict__ vbuf,
    const float* __restrict__ bk0, const float* __restrict__ bv0,
    u16t* __restrict__ ctxkh, float* __restrict__ ctxv,
    float* __restrict__ normacc, int t)
{
  int b = blockIdx.z;
  if (blockIdx.x == 0 && threadIdx.x == 0) normacc[b] = 0.0f;
  int idx = blockIdx.x * 256 + threadIdx.x;      // < 192*1024
  int r = idx >> 10, d = idx & 1023;
  int tok = t * CHK - WIN + r;
  size_t o = ((size_t)b * 192 + r) * DD + d;
  float kv, vv;
  if (tok < 0) {
    size_t i0 = ((size_t)b * WIN + (WIN + tok)) * DD + d;
    kv = bk0[i0]; vv = bv0[i0];
  } else {
    size_t i1 = ((size_t)b * SS + tok) * DD + d;
    kv = kbuf[i1]; vv = vbuf[i1];
  }
  ctxkh[o] = f2bf(kv);
  ctxv[o] = vv;
}

__global__ __launch_bounds__(256) void sumsq_k(
    const float* __restrict__ X, float* __restrict__ acc)
{
  __shared__ float red[256];
  int b = blockIdx.z;
  const float* p = X + (size_t)b * DD * DD;
  float s = 0.0f;
  for (int i = blockIdx.x * 256 + threadIdx.x; i < DD * DD; i += gridDim.x * 256) {
    float v = p[i]; s += v * v;
  }
  red[threadIdx.x] = s; __syncthreads();
  for (int o = 128; o > 0; o >>= 1) {
    if (threadIdx.x < o) red[threadIdx.x] += red[threadIdx.x + o];
    __syncthreads();
  }
  if (threadIdx.x == 0) atomicAdd(&acc[b], red[0]);
}

__global__ __launch_bounds__(256) void ns_scale(
    float* __restrict__ X, u16t* __restrict__ Xh, const float* __restrict__ acc)
{
  int b = blockIdx.z;
  float inv = 1.0f / (sqrtf(acc[b]) + 1e-7f);
  size_t base = (size_t)b * DD * DD + (size_t)blockIdx.x * 1024 + (size_t)threadIdx.x * 4;
#pragma unroll
  for (int i = 0; i < 4; ++i) {
    float v = X[base + i] * inv;
    X[base + i] = v;
    Xh[base + i] = f2bf(v);
  }
}

__global__ __launch_bounds__(256) void update_m(
    float* __restrict__ Mst, u16t* __restrict__ Mh, const float* __restrict__ og,
    const float* __restrict__ etam, const float* __restrict__ alm, int t)
{
  int b = blockIdx.z;
  size_t e = (size_t)blockIdx.x * 1024 + (size_t)threadIdx.x * 4;
  size_t base = (size_t)b * DD * DD + e;
  size_t gbase = ((size_t)b * NCH + t) * DD;
#pragma unroll
  for (int i = 0; i < 4; ++i) {
    int j = (int)((e + i) & 1023);
    float nv = Mst[base + i] * alm[gbase + j] - og[base + i] * etam[gbase + j];
    Mst[base + i] = nv;
    Mh[base + i] = f2bf(nv);
  }
}

__global__ __launch_bounds__(256) void finalize_k(
    float* __restrict__ out, const float* __restrict__ w,
    const float* __restrict__ byp, const float* __restrict__ gam)
{
  __shared__ float red[256];
  size_t row = blockIdx.x;
  float* orow = out + row * DD;
  const float* brow = byp + row * DD;
  const float* grow = gam + row * DD;
  float v[4], s = 0.0f;
#pragma unroll
  for (int i = 0; i < 4; ++i) { v[i] = orow[threadIdx.x + i * 256]; s += v[i] * v[i]; }
  red[threadIdx.x] = s; __syncthreads();
  for (int o = 128; o > 0; o >>= 1) {
    if (threadIdx.x < o) red[threadIdx.x] += red[threadIdx.x + o];
    __syncthreads();
  }
  float inv = rsqrtf(red[0] * (1.0f / DD) + 1e-6f);
#pragma unroll
  for (int i = 0; i < 4; ++i) {
    int d = threadIdx.x + i * 256;
    orow[d] = v[i] * inv * w[d] * brow[d] * grow[d];
  }
}

__global__ __launch_bounds__(256) void tails_k(
    const float* __restrict__ kbuf, const float* __restrict__ vbuf,
    float* __restrict__ bk, float* __restrict__ bv)
{
  size_t idx = (size_t)blockIdx.x * 256 + threadIdx.x;
  int b = (int)(idx >> 17);
  size_t rd = idx & 131071;
  int r = (int)(rd >> 10), d = (int)(rd & 1023);
  size_t src = ((size_t)b * SS + (SS - WIN + r)) * DD + d;
  bk[idx] = kbuf[src];
  bv[idx] = vbuf[src];
}

// ---------------------------------------------------------------------------
enum GMode { G_PROJ, G_PROJ_SILU, G_BF16OUT, G_GRAD, G_XXT, G_XUP, G_F32OUT };

static void gemm(hipStream_t st, GMode mode, const u16t* A, const u16t* B,
                 const float* Cx, float* D, u16t* Dh,
                 int M, int N, int K, int lda, int ldb, int ldd,
                 long sA, long sB, long sC, long sD, int batches,
                 float alpha, float beta)
{
  dim3 g(N / 128, (M + 127) / 128, batches);
  switch (mode) {
  case G_PROJ:
    gemm_k<false, true, 0, true, false><<<g, 256, 0, st>>>(A, B, Cx, D, Dh, M, N, K, lda, ldb, ldd, sA, sB, sC, sD, alpha, beta); break;
  case G_PROJ_SILU:
    gemm_k<false, true, 1, true, false><<<g, 256, 0, st>>>(A, B, Cx, D, Dh, M, N, K, lda, ldb, ldd, sA, sB, sC, sD, alpha, beta); break;
  case G_BF16OUT:
    gemm_k<false, false, 0, false, true><<<g, 256, 0, st>>>(A, B, Cx, D, Dh, M, N, K, lda, ldb, ldd, sA, sB, sC, sD, alpha, beta); break;
  case G_GRAD:
    gemm_k<true, false, 0, true, false><<<g, 256, 0, st>>>(A, B, Cx, D, Dh, M, N, K, lda, ldb, ldd, sA, sB, sC, sD, alpha, beta); break;
  case G_XXT:
    gemm_k<false, true, 0, true, true><<<g, 256, 0, st>>>(A, B, Cx, D, Dh, M, N, K, lda, ldb, ldd, sA, sB, sC, sD, alpha, beta); break;
  case G_XUP:
    gemm_k<false, false, 0, true, true><<<g, 256, 0, st>>>(A, B, Cx, D, Dh, M, N, K, lda, ldb, ldd, sA, sB, sC, sD, alpha, beta); break;
  case G_F32OUT:
    gemm_k<false, false, 0, true, false><<<g, 256, 0, st>>>(A, B, Cx, D, Dh, M, N, K, lda, ldb, ldd, sA, sB, sC, sD, alpha, beta); break;
  }
}

extern "C" void kernel_launch(void* const* d_in, const int* in_sizes, int n_in,
                              void* d_out, int out_size, void* d_ws, size_t ws_size,
                              hipStream_t stream)
{
  (void)in_sizes; (void)n_in; (void)out_size; (void)ws_size;
  const float* x    = (const float*)d_in[0];
  const float* mem0 = (const float*)d_in[1];
  const float* bk0  = (const float*)d_in[2];
  const float* bv0  = (const float*)d_in[3];
  const float* w_in = (const float*)d_in[4];
  const float* w_kq = (const float*)d_in[5];
  const float* w_out= (const float*)d_in[6];
  const float* Wk   = (const float*)d_in[7];
  const float* Wq   = (const float*)d_in[8];
  const float* Wv   = (const float*)d_in[9];
  const float* Wg   = (const float*)d_in[10];
  const float* Wb   = (const float*)d_in[11];
  const float* ckw  = (const float*)d_in[12];
  const float* ckb  = (const float*)d_in[13];
  const float* cqw  = (const float*)d_in[14];
  const float* cqb  = (const float*)d_in[15];

  const size_t BSD  = (size_t)BB * SS * DD;
  const size_t BDD  = (size_t)BB * DD * DD;
  const size_t BCTX = (size_t)BB * 192 * DD;
  const size_t DXD  = (size_t)DD * DD;

  // ---- workspace carve: f32 region then bf16 region ----
  float* kraw   = (float*)d_ws;        // BSD (also eta/alpha scratch)
  float* qraw   = kraw + BSD;          // BSD
  float* kbuf   = qraw + BSD;          // BSD
  float* vbuf   = kbuf + BSD;          // BSD
  float* gamma  = vbuf + BSD;          // BSD
  float* bypass = gamma + BSD;         // BSD
  float* ctxv   = bypass + BSD;        // BCTX
  float* X0     = ctxv + BCTX;         // BDD
  float* X1     = X0 + BDD;            // BDD
  float* Abuf   = X1 + BDD;            // BDD
  float* etam   = Abuf + BDD;          // B*NCH*D
  float* alm    = etam + (size_t)BB * NCH * DD;
  float* normacc= alm + (size_t)BB * NCH * DD;   // 4
  u16t* hb = (u16t*)(((uintptr_t)(normacc + 4) + 15) & ~(uintptr_t)15);
  u16t* xnh   = hb;                    // BSD
  u16t* qh    = xnh + BSD;             // BSD
  u16t* ctxkh = qh + BSD;              // BCTX
  u16t* errh  = ctxkh + BCTX;          // BCTX
  u16t* X0h   = errh + BCTX;           // BDD
  u16t* X1h   = X0h + BDD;             // BDD
  u16t* Ah    = X1h + BDD;             // BDD
  u16t* Bmh   = Ah + BDD;              // BDD
  u16t* Mh    = Bmh + BDD;             // BDD
  u16t* Whk   = Mh + BDD;              // DxD
  u16t* Whq   = Whk + DXD;
  u16t* Whv   = Whq + DXD;
  u16t* Whb   = Whv + DXD;
  u16t* Whg   = Whb + DXD;             // 3*DxD

  float* outp = (float*)d_out;         // B*S*D
  float* Mst  = outp + BSD;            // B*D*D working state
  float* bk   = Mst + BDD;
  float* bv   = bk + (size_t)BB * WIN * DD;

  // M <- mem_state ; bf16 mirrors of M and weights
  hipMemcpyAsync(Mst, mem0, BDD * sizeof(float), hipMemcpyDeviceToDevice, stream);
  cvt_k<<<(int)(BDD / 256), 256, 0, stream>>>(Mst, Mh, (int)BDD);
  cvt_k<<<(int)(DXD / 256), 256, 0, stream>>>(Wk, Whk, (int)DXD);
  cvt_k<<<(int)(DXD / 256), 256, 0, stream>>>(Wq, Whq, (int)DXD);
  cvt_k<<<(int)(DXD / 256), 256, 0, stream>>>(Wv, Whv, (int)DXD);
  cvt_k<<<(int)(DXD / 256), 256, 0, stream>>>(Wb, Whb, (int)DXD);
  cvt_k<<<(int)(3 * DXD / 256), 256, 0, stream>>>(Wg, Whg, (int)(3 * DXD));

  rmsnorm_rows<<<BB * SS, 256, 0, stream>>>(x, w_in, xnh);

  const int MF = BB * SS;
  gemm(stream, G_PROJ,      xnh, Whk, nullptr, kraw,   nullptr, MF, DD, DD, DD, DD, DD, 0,0,0,0, 1, 1.f, 0.f);
  gemm(stream, G_PROJ,      xnh, Whq, nullptr, qraw,   nullptr, MF, DD, DD, DD, DD, DD, 0,0,0,0, 1, 1.f, 0.f);
  gemm(stream, G_PROJ_SILU, xnh, Whv, nullptr, vbuf,   nullptr, MF, DD, DD, DD, DD, DD, 0,0,0,0, 1, 1.f, 0.f);
  gemm(stream, G_PROJ_SILU, xnh, Whb, nullptr, bypass, nullptr, MF, DD, DD, DD, DD, DD, 0,0,0,0, 1, 1.f, 0.f);
  gemm(stream, G_PROJ_SILU, xnh, Whg, nullptr, gamma,  nullptr, MF, DD, DD, DD, DD, DD, 0,0,0,0, 1, 1.f, 0.f);

  {
    dim3 g(SS, BB);
    conv_silu_rms<<<g, 256, 0, stream>>>(kraw, ckw, ckb, w_kq, kbuf, nullptr);
    conv_silu_rms<<<g, 256, 0, stream>>>(qraw, cqw, cqb, w_kq, nullptr, qh);
  }

  gemm(stream, G_PROJ_SILU, xnh, Whg + DXD,     nullptr, kraw, nullptr, MF, DD, DD, DD, DD, DD, 0,0,0,0, 1, 1.f, 0.f);
  chunk_mean<<<dim3(BB * NCH, DD / 256), 256, 0, stream>>>(kraw, etam);
  gemm(stream, G_PROJ_SILU, xnh, Whg + 2 * DXD, nullptr, kraw, nullptr, MF, DD, DD, DD, DD, DD, 0,0,0,0, 1, 1.f, 0.f);
  chunk_mean<<<dim3(BB * NCH, DD / 256), 256, 0, stream>>>(kraw, alm);

  const long sCtx = 192 * DD, sDxD = (long)DXD, sSeq = (long)SS * DD;
  const float a_ns = 3.4445f, b_ns = -4.775f, c_ns = 2.0315f;

  for (int t = 0; t < NCH; ++t) {
    build_ctx<<<dim3(768, 1, BB), 256, 0, stream>>>(kbuf, vbuf, bk0, bv0, ctxkh, ctxv, normacc, t);

    // err(bf16) = ctx_k @ M - ctx_v
    gemm(stream, G_BF16OUT, ctxkh, Mh, ctxv, nullptr, errh, 192, DD, DD, DD, DD, DD,
         sCtx, sDxD, sCtx, sCtx, BB, 1.f, -1.f);
    // grad(f32) = ctx_k^T @ err
    gemm(stream, G_GRAD, ctxkh, errh, nullptr, X0, nullptr, DD, DD, 192, DD, DD, DD,
         sCtx, sCtx, 0, sDxD, BB, 1.f, 0.f);

    sumsq_k<<<dim3(512, 1, BB), 256, 0, stream>>>(X0, normacc);
    ns_scale<<<dim3(1024, 1, BB), 256, 0, stream>>>(X0, X0h, normacc);

    float* Xc = X0; u16t* Xch = X0h;
    float* Xa = X1; u16t* Xah = X1h;
    for (int it = 0; it < 5; ++it) {
      gemm(stream, G_XXT, Xch, Xch, nullptr, Abuf, Ah, DD, DD, DD, DD, DD, DD,
           sDxD, sDxD, 0, sDxD, BB, 1.f, 0.f);                       // A = X@X^T
      gemm(stream, G_BF16OUT, Ah, Ah, Abuf, nullptr, Bmh, DD, DD, DD, DD, DD, DD,
           sDxD, sDxD, sDxD, sDxD, BB, c_ns, b_ns);                  // Bm = c*(A@A)+b*A
      gemm(stream, G_XUP, Bmh, Xch, Xc, Xa, Xah, DD, DD, DD, DD, DD, DD,
           sDxD, sDxD, sDxD, sDxD, BB, 1.f, a_ns);                   // X' = Bm@X + a*X
      { float* tf = Xc; Xc = Xa; Xa = tf; }
      { u16t* th = Xch; Xch = Xah; Xah = th; }
    }

    update_m<<<dim3(1024, 1, BB), 256, 0, stream>>>(Mst, Mh, Xc, etam, alm, t);

    gemm(stream, G_F32OUT, qh + (size_t)t * CHK * DD, Mh, nullptr,
         outp + (size_t)t * CHK * DD, nullptr,
         CHK, DD, DD, DD, DD, DD, sSeq, sDxD, 0, sSeq, BB, 1.f, 0.f);
  }

  finalize_k<<<BB * SS, 256, 0, stream>>>(outp, w_out, bypass, gamma);
  tails_k<<<(BB * WIN * DD) / 256, 256, 0, stream>>>(kbuf, vbuf, bk, bv);
}
